// CellConv_57939108823649
// MI455X (gfx1250) — matrix operand order repaired and checked by reading, not compiled, not run on hardware
//
#include <hip/hip_runtime.h>
#include <hip/hip_bf16.h>

typedef _Float16 half_t;
typedef __attribute__((ext_vector_type(8)))  _Float16 v8h;
typedef __attribute__((ext_vector_type(16))) _Float16 v16h;
typedef __attribute__((ext_vector_type(8)))  float    v8f;

#define E_TOTAL 400000
#define WAVES 4
#define BLOCK (WAVES * 32)
#define EDGES_PER_WAVE 16
#define EDGES_PER_BLOCK (WAVES * EDGES_PER_WAVE)

// ---------------------------------------------------------------------------
// LDS layout: weights (shared by block) + per-wave activation slabs.
// Weights stored [out_pad][K_pad] in f16 so a B-fragment is 2x ds_load_b128.
// Activations stored [M][K_pad] in f16 so an A-fragment is 2x ds_load_b128.
// ---------------------------------------------------------------------------
struct __align__(16) Smem {
  half_t wq1[16 * 192];           // q layer1: 180->16 (Kpad 192)
  half_t ws1[16 * 224];           // scale layer1: 196->16 (Kpad 224)
  half_t wq2[16 * 32], wq3[32 * 32];
  half_t wa1[16 * 32], wa2[16 * 32], wa3[16 * 32];
  half_t wi1[16 * 32], wi2[16 * 32], wi3[16 * 32];
  half_t ws2[16 * 32], ws3[16 * 32];
  half_t xin[WAVES][16 * 224];    // concat(last_nf,src_nf,dst_nf,r,r_in,0pad)
  half_t hA[WAVES][16 * 32];      // hidden ping
  half_t hB[WAVES][16 * 32];      // hidden pong
  half_t qbuf[WAVES][16 * 32];    // q outputs (24 used)
  half_t abuf[WAVES][16 * 32];    // att/att_in input staging (17/9 used)
  float  obuf[WAVES][16 * 16];    // att layer3 output, f32
  float  rbuf[WAVES][16 * 8];
  float  ribuf[WAVES][16 * 8];
  float  sbuf[WAVES][16 * 2];
  float  bq1[16], bq2[16], bq3[32];
  float  ba1[16], ba2[16], ba3[16];
  float  bi1[16], bi2[16], bi3[16];
  float  bs1[16], bs2[16], bs3[16];
  int    sidx[WAVES][16], didx[WAVES][16];
};

__device__ __forceinline__ float sigmoidf(float x) {
  return 1.0f / (1.0f + __expf(-x));
}

// A fragment: 16x32 f16, rows M=0..15 striped per ISA 7.12.2 (lanes 0-15 hold
// K 0-7 / 16-23 in v0-7; lanes 16-31 hold K 8-15 / 24-31).
__device__ __forceinline__ v16h afrag(const half_t* X, int kpad, int kbase, int lane) {
  int m = lane & 15, hi = lane >> 4;
  const half_t* p = X + m * kpad + kbase + hi * 8;
  v8h lo = *(const v8h*)p;
  v8h hh = *(const v8h*)(p + 16);
  return __builtin_shufflevector(lo, hh, 0,1,2,3,4,5,6,7,8,9,10,11,12,13,14,15);
}

// B fragment: 32x16 f16, col N = lane&15, K = 8*(lane>=16) + 0..7 / 16 + 8*(lane>=16) + 0..7.
// W stored as [N][Kpad] so per-lane K run is contiguous.
__device__ __forceinline__ v16h bfrag(const half_t* W, int kpad, int kbase, int lane) {
  int n = lane & 15, hi = lane >> 4;
  const half_t* p = W + n * kpad + kbase + hi * 8;
  v8h lo = *(const v8h*)p;
  v8h hh = *(const v8h*)(p + 16);
  return __builtin_shufflevector(lo, hh, 0,1,2,3,4,5,6,7,8,9,10,11,12,13,14,15);
}

__device__ __forceinline__ v8f mlp_layer(const half_t* X, int xkpad,
                                         const half_t* W, int wkpad,
                                         int nchunks, const float* bias, int lane) {
  v8f c = {};
  for (int kc = 0; kc < nchunks; ++kc) {
    v16h a = afrag(X, xkpad, kc * 32, lane);
    v16h b = bfrag(W, wkpad, kc * 32, lane);
    c = __builtin_amdgcn_wmma_f32_16x16x32_f16(false, a, false, b,
                                               (short)0, c, false, false);
  }
  float bv = bias[lane & 15];
#pragma unroll
  for (int v = 0; v < 8; ++v) c[v] += bv;
  return c;
}

// C/D layout: VGPR v holds M = v + 8*(lane>=16), N = lane&15.
__device__ __forceinline__ void store_h(half_t* Hd, int npad, v8f c, bool lrelu, int lane) {
  int n = lane & 15, hi = lane >> 4;
#pragma unroll
  for (int v = 0; v < 8; ++v) {
    float x = c[v];
    if (lrelu) x = x > 0.0f ? x : 0.2f * x;
    Hd[(v + 8 * hi) * npad + n] = (half_t)x;
  }
}

__device__ __forceinline__ void store_f(float* Od, int npad, v8f c, int lane) {
  int n = lane & 15, hi = lane >> 4;
#pragma unroll
  for (int v = 0; v < 8; ++v) Od[(v + 8 * hi) * npad + n] = c[v];
}

__device__ __forceinline__ void loadW(half_t* d, const float* g, int out, int in,
                                      int npad, int kpad) {
  for (int idx = threadIdx.x; idx < npad * kpad; idx += BLOCK) {
    int n = idx / kpad, k = idx - n * kpad;
    d[idx] = (half_t)((n < out && k < in) ? g[n * in + k] : 0.0f);
  }
}
__device__ __forceinline__ void loadB(float* d, const float* g, int out, int npad) {
  for (int idx = threadIdx.x; idx < npad; idx += BLOCK)
    d[idx] = (idx < out) ? g[idx] : 0.0f;
}

__global__ __launch_bounds__(BLOCK) void cellconv_kernel(
    const float* __restrict__ nin, const float* __restrict__ nf,
    const float* __restrict__ ef, const float* __restrict__ efin,
    const int* __restrict__ src, const int* __restrict__ dst,
    const float* qW1, const float* qb1, const float* qW2, const float* qb2,
    const float* qW3, const float* qb3,
    const float* aW1, const float* ab1, const float* aW2, const float* ab2,
    const float* aW3, const float* ab3,
    const float* iW1, const float* ib1, const float* iW2, const float* ib2,
    const float* iW3, const float* ib3,
    const float* sW1, const float* sb1, const float* sW2, const float* sb2,
    const float* sW3, const float* sb3,
    float* __restrict__ out) {
  __shared__ Smem sm;
  const int tid = threadIdx.x;
  const int lane = tid & 31;
  const int w = tid >> 5;

  // ---- stage all MLP weights into LDS (f16, padded) ----
  loadW(sm.wq1, qW1, 16, 180, 16, 192);
  loadW(sm.wq2, qW2, 16, 16, 16, 32);
  loadW(sm.wq3, qW3, 24, 16, 32, 32);
  loadW(sm.wa1, aW1, 16, 17, 16, 32);
  loadW(sm.wa2, aW2, 16, 16, 16, 32);
  loadW(sm.wa3, aW3, 14, 16, 16, 32);
  loadW(sm.wi1, iW1, 16,  9, 16, 32);
  loadW(sm.wi2, iW2, 16, 16, 16, 32);
  loadW(sm.wi3, iW3,  7, 16, 16, 32);
  loadW(sm.ws1, sW1, 16, 196, 16, 224);
  loadW(sm.ws2, sW2, 16, 16, 16, 32);
  loadW(sm.ws3, sW3,  2, 16, 16, 32);
  loadB(sm.bq1, qb1, 16, 16); loadB(sm.bq2, qb2, 16, 16); loadB(sm.bq3, qb3, 24, 32);
  loadB(sm.ba1, ab1, 16, 16); loadB(sm.ba2, ab2, 16, 16); loadB(sm.ba3, ab3, 14, 16);
  loadB(sm.bi1, ib1, 16, 16); loadB(sm.bi2, ib2, 16, 16); loadB(sm.bi3, ib3,  7, 16);
  loadB(sm.bs1, sb1, 16, 16); loadB(sm.bs2, sb2, 16, 16); loadB(sm.bs3, sb3,  2, 16);

  const int e0 = blockIdx.x * EDGES_PER_BLOCK + w * EDGES_PER_WAVE;
  if (lane < 16) sm.sidx[w][lane] = src[e0 + lane];
  else           sm.didx[w][lane - 16] = dst[e0 + lane - 16];
  // zero hidden pads (cols 16..31 never rewritten by store_h)
  for (int idx = lane; idx < 16 * 32; idx += 32) {
    sm.hA[w][idx] = (half_t)0.0f;
    sm.hB[w][idx] = (half_t)0.0f;
  }
  __syncthreads();

  // ---- gather concat(last_nf, src_nf, dst_nf) into Xin; cols >=180 zero ----
  half_t* X = sm.xin[w];
  for (int idx = lane; idx < 16 * 224; idx += 32) {
    int le = idx / 224, c = idx - le * 224;
    float v = 0.0f;
    if (c < 52)       v = nf [(size_t)sm.sidx[w][le] * 52 + c];
    else if (c < 116) v = nin[(size_t)sm.sidx[w][le] * 64 + (c - 52)];
    else if (c < 180) v = nin[(size_t)sm.didx[w][le] * 64 + (c - 116)];
    X[idx] = (half_t)v;
  }
  __syncthreads();

  // ---- q MLP: 180 -> 16 -> 16 -> 24 ----
  v8f c = mlp_layer(X, 224, sm.wq1, 192, 6, sm.bq1, lane);
  store_h(sm.hA[w], 32, c, true, lane);
  __syncthreads();
  c = mlp_layer(sm.hA[w], 32, sm.wq2, 32, 1, sm.bq2, lane);
  store_h(sm.hB[w], 32, c, true, lane);
  __syncthreads();
  c = mlp_layer(sm.hB[w], 32, sm.wq3, 32, 1, sm.bq3, lane);
  store_h(sm.qbuf[w], 32, c, false, lane);
  c = mlp_layer(sm.hB[w], 32, sm.wq3 + 16 * 32, 32, 1, sm.bq3 + 16, lane);
  store_h(sm.qbuf[w] + 16, 32, c, false, lane);
  __syncthreads();

  // ---- att MLP over 8 subtiles of 16 rows (row = edge*8 + n*2 + d) ----
  for (int t = 0; t < 8; ++t) {
    for (int idx = lane; idx < 16 * 32; idx += 32) {
      int i = idx >> 5, cc = idx & 31;
      int le = 2 * t + (i >> 3), j = i & 7, nh = j >> 1;
      half_t v = (half_t)0.0f;
      if (cc < 2)       v = sm.qbuf[w][le * 32 + 3 * j + cc];
      else if (cc < 17) v = (half_t)ef[(size_t)(e0 + le) * 256 + nh * 15 + (cc - 2)];
      sm.abuf[w][idx] = v;
    }
    __syncthreads();
    v8f h = mlp_layer(sm.abuf[w], 32, sm.wa1, 32, 1, sm.ba1, lane);
    store_h(sm.hA[w], 32, h, true, lane);
    __syncthreads();
    h = mlp_layer(sm.hA[w], 32, sm.wa2, 32, 1, sm.ba2, lane);
    store_h(sm.hB[w], 32, h, true, lane);
    __syncthreads();
    h = mlp_layer(sm.hB[w], 32, sm.wa3, 32, 1, sm.ba3, lane);
    store_f(sm.obuf[w], 16, h, lane);
    __syncthreads();
    if (lane < 16) {
      int i = lane, le = 2 * t + (i >> 3), j = i & 7, nh = j >> 1;
      size_t e = (size_t)(e0 + le);
      const float* tab = ef + e * 256 + 60 + nh * 49;
      float ax[7], ay[7];
#pragma unroll
      for (int s = 0; s < 7; ++s) {
        ax[s] = sm.obuf[w][i * 16 + s];
        ay[s] = sm.obuf[w][i * 16 + 7 + s];
      }
      float ssum = 0.0f, acc = 0.0f;
#pragma unroll
      for (int s1 = 0; s1 < 7; ++s1)
#pragma unroll
        for (int s2 = 0; s2 < 7; ++s2) {
          float o = sigmoidf(ax[s1] * ay[s2]);
          ssum += o;
          acc += tab[s1 * 7 + s2] * o;
        }
      float r = acc / (ssum + 1e-10f);
      sm.rbuf[w][le * 8 + j] = r;
      sm.xin[w][le * 224 + 180 + j] = (half_t)r;
    }
    __syncthreads();
  }

  // ---- att_in MLP over 8 subtiles ----
  for (int t = 0; t < 8; ++t) {
    for (int idx = lane; idx < 16 * 32; idx += 32) {
      int i = idx >> 5, cc = idx & 31;
      int le = 2 * t + (i >> 3), j = i & 7, nh = j >> 1;
      half_t v = (half_t)0.0f;
      if (cc == 0)     v = sm.qbuf[w][le * 32 + 3 * j + 2];
      else if (cc < 9) v = (half_t)efin[(size_t)(e0 + le) * 60 + nh * 8 + (cc - 1)];
      sm.abuf[w][idx] = v;
    }
    __syncthreads();
    v8f h = mlp_layer(sm.abuf[w], 32, sm.wi1, 32, 1, sm.bi1, lane);
    store_h(sm.hA[w], 32, h, true, lane);
    __syncthreads();
    h = mlp_layer(sm.hA[w], 32, sm.wi2, 32, 1, sm.bi2, lane);
    store_h(sm.hB[w], 32, h, true, lane);
    __syncthreads();
    h = mlp_layer(sm.hB[w], 32, sm.wi3, 32, 1, sm.bi3, lane);
    store_f(sm.obuf[w], 16, h, lane);
    __syncthreads();
    if (lane < 16) {
      int i = lane, le = 2 * t + (i >> 3), j = i & 7, nh = j >> 1;
      size_t e = (size_t)(e0 + le);
      const float* tab = efin + e * 60 + 32 + nh * 7;
      float ssum = 0.0f, acc = 0.0f;
#pragma unroll
      for (int s = 0; s < 7; ++s) {
        float o = sigmoidf(sm.obuf[w][i * 16 + s]);
        ssum += o;
        acc += tab[s] * o;
      }
      float rin = acc / (ssum + 1e-10f);
      sm.ribuf[w][le * 8 + j] = rin;
      sm.xin[w][le * 224 + 188 + j] = (half_t)rin;
    }
    __syncthreads();
  }

  // ---- scale MLP: 196 -> 16 -> 16 -> 2 ----
  c = mlp_layer(X, 224, sm.ws1, 224, 7, sm.bs1, lane);
  store_h(sm.hA[w], 32, c, true, lane);
  __syncthreads();
  c = mlp_layer(sm.hA[w], 32, sm.ws2, 32, 1, sm.bs2, lane);
  store_h(sm.hB[w], 32, c, true, lane);
  __syncthreads();
  c = mlp_layer(sm.hB[w], 32, sm.ws3, 32, 1, sm.bs3, lane);
  {
    int n = lane & 15, hi = lane >> 4;
    if (n < 2) {
#pragma unroll
      for (int v = 0; v < 8; ++v) sm.sbuf[w][(v + 8 * hi) * 2 + n] = c[v];
    }
  }
  __syncthreads();

  // ---- final reduction + output ----
  if (lane < 16) {
    int le = lane;
    size_t e = (size_t)(e0 + le);
    float mr = 0.0f, mi = 0.0f;
#pragma unroll
    for (int k = 0; k < 8; ++k) {
      mr += sm.rbuf[w][le * 8 + k];
      mi += sm.ribuf[w][le * 8 + k];
    }
    mr *= 0.125f;
    mi *= 0.125f;
    out[e]           = mr * (0.9f + 0.2f * sigmoidf(sm.sbuf[w][le * 2 + 0]));
    out[E_TOTAL + e] = mi * (0.9f + 0.2f * sigmoidf(sm.sbuf[w][le * 2 + 1]));
  }
}

extern "C" void kernel_launch(void* const* d_in, const int* in_sizes, int n_in,
                              void* d_out, int out_size, void* d_ws, size_t ws_size,
                              hipStream_t stream) {
  (void)in_sizes; (void)n_in; (void)out_size; (void)d_ws; (void)ws_size;
  const float* nin  = (const float*)d_in[0];
  const float* nf   = (const float*)d_in[1];
  const float* ef   = (const float*)d_in[2];
  const float* efin = (const float*)d_in[3];
  const int*   src  = (const int*)d_in[4];
  const int*   dst  = (const int*)d_in[5];
  const float* qW1 = (const float*)d_in[6],  *qb1 = (const float*)d_in[7];
  const float* qW2 = (const float*)d_in[8],  *qb2 = (const float*)d_in[9];
  const float* qW3 = (const float*)d_in[10], *qb3 = (const float*)d_in[11];
  const float* aW1 = (const float*)d_in[12], *ab1 = (const float*)d_in[13];
  const float* aW2 = (const float*)d_in[14], *ab2 = (const float*)d_in[15];
  const float* aW3 = (const float*)d_in[16], *ab3 = (const float*)d_in[17];
  const float* iW1 = (const float*)d_in[18], *ib1 = (const float*)d_in[19];
  const float* iW2 = (const float*)d_in[20], *ib2 = (const float*)d_in[21];
  const float* iW3 = (const float*)d_in[22], *ib3 = (const float*)d_in[23];
  const float* sW1 = (const float*)d_in[24], *sb1 = (const float*)d_in[25];
  const float* sW2 = (const float*)d_in[26], *sb2 = (const float*)d_in[27];
  const float* sW3 = (const float*)d_in[28], *sb3 = (const float*)d_in[29];
  float* out = (float*)d_out;

  dim3 grid(E_TOTAL / EDGES_PER_BLOCK);  // 6250 blocks, exact
  dim3 block(BLOCK);
  cellconv_kernel<<<grid, block, 0, stream>>>(
      nin, nf, ef, efin, src, dst,
      qW1, qb1, qW2, qb2, qW3, qb3,
      aW1, ab1, aW2, ab2, aW3, ab3,
      iW1, ib1, iW2, ib2, iW3, ib3,
      sW1, sb1, sW2, sb2, sW3, sb3,
      out);
}
